// RNNMultiStepModule_5600637354611
// MI455X (gfx1250) — matrix operand order repaired and checked by reading, not compiled, not run on hardware
//
#include <hip/hip_runtime.h>
#include <stdint.h>

#define BATCH   8192
#define H       1024
#define K3      3072   // 3*H
#define NSTEPS  64

typedef __attribute__((ext_vector_type(16))) __bf16        v16bf;
typedef __attribute__((ext_vector_type(8)))  float         v8f;
typedef __attribute__((ext_vector_type(4)))  unsigned int  v4u;
typedef __attribute__((ext_vector_type(8)))  unsigned int  v8u;

static __device__ __forceinline__ unsigned short f2bf(float f) {
    unsigned int u = __builtin_bit_cast(unsigned int, f);
    u += 0x7FFFu + ((u >> 16) & 1u);   // round-to-nearest-even
    return (unsigned short)(u >> 16);
}

// sigmoid via v_exp_f32 + v_rcp_f32 (avoid IEEE div expansion)
static __device__ __forceinline__ float sigm(float x) {
    return __builtin_amdgcn_rcpf(1.0f + __expf(-x));
}

// tanh: native v_tanh_f32 on gfx1250 if the toolchain exposes it,
// else rcp-based identity tanh(x) = 1 - 2/(exp(2x)+1)
static __device__ __forceinline__ float fast_tanh(float x) {
#if defined(__has_builtin)
#if __has_builtin(__builtin_amdgcn_tanhf)
    return __builtin_amdgcn_tanhf(x);
#else
    return 1.0f - 2.0f * __builtin_amdgcn_rcpf(__expf(2.0f * x) + 1.0f);
#endif
#else
    return 1.0f - 2.0f * __builtin_amdgcn_rcpf(__expf(2.0f * x) + 1.0f);
#endif
}

// A-fragment: 16x32 bf16 tile of h. Lanes 0-15 -> M=lane, K=k0..k0+15;
// lanes 16-31 -> same M, K=k0+16..k0+31 (matches B-fragment k order).
// => two 16B loads, contiguous.
static __device__ __forceinline__ v16bf loadA(const unsigned short* p) {
    v4u lo = *(const v4u*)p;
    v4u hi = *(const v4u*)(p + 8);
    v8u t;
    t[0]=lo[0]; t[1]=lo[1]; t[2]=lo[2]; t[3]=lo[3];
    t[4]=hi[0]; t[5]=hi[1]; t[6]=hi[2]; t[7]=hi[3];
    return __builtin_bit_cast(v16bf, t);
}

// B-fragment: 32x16 bf16, B[k,n] = W[jbase+n, k]. Lane n (0-15) holds K=k0..k0+15,
// lanes 16-31 hold K=k0+16..k0+31 -> 32 contiguous bytes of row-major W. 32B aligned.
static __device__ __forceinline__ v16bf loadB(const unsigned short* p) {
    v4u lo = *(const v4u*)p;
    v4u hi = *(const v4u*)(p + 8);
    v8u t;
    t[0]=lo[0]; t[1]=lo[1]; t[2]=lo[2]; t[3]=lo[3];
    t[4]=hi[0]; t[5]=hi[1]; t[6]=hi[2]; t[7]=hi[3];
    return __builtin_bit_cast(v16bf, t);
}

// ---------------- prep kernels ----------------

__global__ __launch_bounds__(256)
void convert_to_bf16(const float* __restrict__ src, unsigned short* __restrict__ dst, int n4) {
    int i = blockIdx.x * 256 + threadIdx.x;
    if (i < n4) {
        float4 f = ((const float4*)src)[i];
        ushort4 o;
        o.x = f2bf(f.x); o.y = f2bf(f.y); o.z = f2bf(f.z); o.w = f2bf(f.w);
        ((ushort4*)dst)[i] = o;
    }
}

// bias[t][j] = (t+1)*W_ih[j] + b_ih[j] + (j < 2H ? b_hh[j] : 0)
// (n-gate keeps b_hh_n separate: n = tanh(gi_n + r*(acc_n + b_hh_n)))
__global__ __launch_bounds__(256)
void make_bias(const float* __restrict__ Wih, const float* __restrict__ bih,
               const float* __restrict__ bhh, float* __restrict__ bias) {
    int idx = blockIdx.x * 256 + threadIdx.x;
    if (idx < NSTEPS * K3) {
        int t = idx / K3;
        int j = idx - t * K3;
        float v = (float)(t + 1) * Wih[j] + bih[j];
        if (j < 2 * H) v += bhh[j];
        bias[idx] = v;
    }
}

// ---------------- fused GEMM + GRU gate step ----------------
// Block: 256 threads = 8 waves, tile BM=128 x BN=64.
// Wave: 64(M) x 16(N) -> 4 M-subtiles, 3 gate accumulators each.
__global__ __launch_bounds__(256)
void gru_step(const unsigned short* __restrict__ hbf_src,
              const float*          __restrict__ hf_src,
              const float*          __restrict__ bias_t,   // 3H for this step
              const float*          __restrict__ bhh,      // raw b_hh (3H)
              const unsigned short* __restrict__ Wbf,      // 3H x H bf16 row-major
              float*                __restrict__ hf_dst,
              unsigned short*       __restrict__ hbf_dst)
{
    const int lane = threadIdx.x & 31;
    const int wave = threadIdx.x >> 5;
    const int mw   = wave >> 2;            // 0..1
    const int nw   = wave & 3;             // 0..3
    const int m0   = blockIdx.y * 128 + mw * 64;
    const int n0   = blockIdx.x * 64 + nw * 16;
    const int l15  = lane & 15;
    const int hi   = lane >> 4;            // half-wave select

    v8f accR[4], accZ[4], accN[4];
#pragma unroll
    for (int s = 0; s < 4; ++s) {
        accR[s] = (v8f){0,0,0,0,0,0,0,0};
        accZ[s] = (v8f){0,0,0,0,0,0,0,0};
        accN[s] = (v8f){0,0,0,0,0,0,0,0};
    }

    const unsigned short* Abase = hbf_src + (size_t)(m0 + l15) * H + hi * 16;
    const unsigned short* B0 = Wbf + (size_t)(0 * H + n0 + l15) * H + hi * 16;
    const unsigned short* B1 = Wbf + (size_t)(1 * H + n0 + l15) * H + hi * 16;
    const unsigned short* B2 = Wbf + (size_t)(2 * H + n0 + l15) * H + hi * 16;

    for (int k0 = 0; k0 < H; k0 += 32) {
        v16bf b0 = loadB(B0 + k0);
        v16bf b1 = loadB(B1 + k0);
        v16bf b2 = loadB(B2 + k0);
#pragma unroll
        for (int s = 0; s < 4; ++s) {
            v16bf a = loadA(Abase + (size_t)s * 16 * H + k0);
            accR[s] = __builtin_amdgcn_wmma_f32_16x16x32_bf16(
                false, a, false, b0, (short)0, accR[s], false, false);
            accZ[s] = __builtin_amdgcn_wmma_f32_16x16x32_bf16(
                false, a, false, b1, (short)0, accZ[s], false, false);
            accN[s] = __builtin_amdgcn_wmma_f32_16x16x32_bf16(
                false, a, false, b2, (short)0, accN[s], false, false);
        }
    }

    const int   col  = n0 + l15;
    const float bR   = bias_t[col];
    const float bZ   = bias_t[H + col];
    const float giN  = bias_t[2 * H + col];
    const float bhhN = bhh[2 * H + col];

#pragma unroll
    for (int s = 0; s < 4; ++s) {
        const int rbase = m0 + s * 16 + hi * 8;
#pragma unroll
        for (int i = 0; i < 8; ++i) {
            const size_t off = (size_t)(rbase + i) * H + col;
            float r    = sigm(accR[s][i] + bR);
            float zg   = sigm(accZ[s][i] + bZ);
            float nn   = fast_tanh(giN + r * (accN[s][i] + bhhN));
            float hold = hf_src[off];
            float hnew = fmaf(zg, hold - nn, nn);   // (1-z)*n + z*h
            hf_dst[off]  = hnew;
            hbf_dst[off] = f2bf(hnew);
        }
    }
}

// ---------------- launch ----------------

static inline size_t align256(size_t x) { return (x + 255) & ~(size_t)255; }

extern "C" void kernel_launch(void* const* d_in, const int* in_sizes, int n_in,
                              void* d_out, int out_size, void* d_ws, size_t ws_size,
                              hipStream_t stream) {
    const float* z   = (const float*)d_in[0];   // (B, H)
    const float* Wih = (const float*)d_in[1];   // (3H, 1)
    const float* Whh = (const float*)d_in[2];   // (3H, H)
    const float* bih = (const float*)d_in[3];   // (3H,)
    const float* bhh = (const float*)d_in[4];   // (3H,)
    (void)in_sizes; (void)n_in; (void)out_size; (void)ws_size;

    char* ws = (char*)d_ws;
    unsigned short* Wbf  = (unsigned short*)ws; ws += align256((size_t)K3 * H * 2);
    unsigned short* hbf0 = (unsigned short*)ws; ws += align256((size_t)BATCH * H * 2);
    unsigned short* hbf1 = (unsigned short*)ws; ws += align256((size_t)BATCH * H * 2);
    float* bias = (float*)ws;                   ws += align256((size_t)NSTEPS * K3 * 4);
    float* hfA  = (float*)ws;                   ws += align256((size_t)BATCH * H * 4);
    float* hfB  = (float*)ws;

    // 1) W_hh -> bf16 (row-major kept: B-fragments load contiguously)
    {
        int n4 = K3 * H / 4;
        convert_to_bf16<<<(n4 + 255) / 256, 256, 0, stream>>>(Whh, Wbf, n4);
    }
    // 2) per-step fused bias vectors
    {
        int n = NSTEPS * K3;
        make_bias<<<(n + 255) / 256, 256, 0, stream>>>(Wih, bih, bhh, bias);
    }
    // 3) h0 = z -> bf16
    {
        int n4 = BATCH * H / 4;
        convert_to_bf16<<<(n4 + 255) / 256, 256, 0, stream>>>(z, hbf0, n4);
    }

    // 4) 64 sequential fused GEMM+gate steps (launch boundary = global sync)
    unsigned short* hbf[2] = { hbf0, hbf1 };
    const float* src_f = z;
    dim3 grid(H / 64, BATCH / 128);   // (16, 64)
    for (int step = 0; step < NSTEPS; ++step) {
        unsigned short* src_b = hbf[step & 1];
        unsigned short* dst_b = hbf[(step + 1) & 1];
        float* dst_f = (step == NSTEPS - 1) ? (float*)d_out
                                            : ((step & 1) ? hfB : hfA);
        gru_step<<<grid, 256, 0, stream>>>(src_b, src_f, bias + (size_t)step * K3,
                                           bhh, Wbf, dst_f, dst_b);
        src_f = dst_f;
    }
}
